// dp_fcn8s_65919158059492
// MI455X (gfx1250) — matrix operand order repaired and checked by reading, not compiled, not run on hardware
//
#include <hip/hip_runtime.h>
#include <hip/hip_bf16.h>
#include <math.h>

// ---------------------------------------------------------------------------
// Types for CDNA5 WMMA
// ---------------------------------------------------------------------------
typedef __bf16 bf16;
typedef __attribute__((ext_vector_type(16))) __bf16 v16bf;
typedef __attribute__((ext_vector_type(8)))  float  v8f;

#define BM 64    // output-channel tile per block
#define BN 128   // pixel tile per block
#define BK 32    // K (Cin*KH*KW) step == WMMA K for bf16

// ---------------------------------------------------------------------------
// f32 -> bf16 conversion (weights / input image)
// ---------------------------------------------------------------------------
__global__ void k_cvt_bf16(const float* __restrict__ s, bf16* __restrict__ d, int n) {
    int i = blockIdx.x * blockDim.x + threadIdx.x;
    if (i < n) d[i] = (bf16)s[i];
}

// ---------------------------------------------------------------------------
// Implicit-GEMM convolution via V_WMMA_F32_16X16X32_BF16.
//   GEMM view: C[M=Cout][N=OH*OW] = sum_K  W[M][K] * im2col(X)[K][N]
//   Block: 256 threads = 8 waves = 2(M) x 4(N); each wave owns a 32x32 tile
//   -> 2 A-fragments x 2 B-fragments -> 4 WMMAs per K-step (register reuse).
// ---------------------------------------------------------------------------
__global__ __launch_bounds__(256)
void k_conv_wmma(const bf16* __restrict__ in, int Cin, int IH, int IW,
                 const bf16* __restrict__ wgt, const float* __restrict__ bias,
                 int Cout, int KH, int KW, int pad,
                 bf16* __restrict__ outB, float* __restrict__ outF,
                 int OH, int OW, int relu)
{
    const int K = Cin * KH * KW;
    const int N = OH * OW;
    const int n0 = blockIdx.x * BN;
    const int m0 = blockIdx.y * BM;

    __shared__ bf16 As[BM][BK + 2];    // weights tile       (64 x 32)
    __shared__ bf16 Bs[BK][BN + 2];    // im2col tile        (32 x 128)

    const int tid  = threadIdx.x;
    const int lane = tid & 31;
    const int wave = tid >> 5;         // 0..7
    const int mw   = wave >> 2;        // 0..1  (M super-tile, 32 rows)
    const int nw   = wave & 3;         // 0..3  (N super-tile, 32 cols)
    const int half = lane >> 4;        // lane half (wave32)
    const int l15  = lane & 15;

    v8f acc[2][2] = {};

    // per-thread staging coordinates
    const int a_row = tid >> 2;              // 0..63
    const int a_col = (tid & 3) << 3;        // 0,8,16,24 (8 bf16 each)
    const int b_kk  = tid >> 3;              // 0..31
    const int b_nn0 = (tid & 7) << 4;        // 16 pixels each

    for (int k0 = 0; k0 < K; k0 += BK) {
        // ---- stage A (weights, OIHW is already [M][K]) ----
        {
            int m = m0 + a_row;
            const bf16* wrow = wgt + (size_t)m * K + k0 + a_col;
            if (m < Cout && k0 + BK < K)
                __builtin_prefetch(wrow + BK, 0, 1);   // global_prefetch next tile
            #pragma unroll
            for (int j = 0; j < 8; ++j) {
                int k = k0 + a_col + j;
                bf16 v = (bf16)0.f;
                if (m < Cout && k < K) v = wgt[(size_t)m * K + k];
                As[a_row][a_col + j] = v;
            }
        }
        // ---- stage B (im2col gather): one (c,r,s) per thread, 16 pixels ----
        {
            int kg = k0 + b_kk;
            int c = 0, r = 0, s = 0, vk = 0;
            if (kg < K) {
                vk = 1;
                c = kg / (KH * KW);
                int rs = kg % (KH * KW);
                r = rs / KW; s = rs % KW;
            }
            const bf16* inc = in + (size_t)c * IH * IW;
            #pragma unroll
            for (int j = 0; j < 16; ++j) {
                int nn = b_nn0 + j;
                int ng = n0 + nn;
                bf16 v = (bf16)0.f;
                if (vk && ng < N) {
                    int oy = ng / OW, ox = ng % OW;
                    int iy = oy + r - pad;
                    int ix = ox + s - pad;
                    if (iy >= 0 && iy < IH && ix >= 0 && ix < IW)
                        v = inc[iy * IW + ix];
                }
                Bs[b_kk][nn] = v;
            }
        }
        __syncthreads();

        // ---- load fragments per CDNA5 16-bit WMMA VGPR layouts ----
        v16bf af[2], bfm[2];
        #pragma unroll
        for (int mi = 0; mi < 2; ++mi) {
            const int am = mw * 32 + mi * 16 + l15;
            #pragma unroll
            for (int e = 0; e < 16; ++e) {
                int v  = e >> 1;
                int kk = ((v < 4) ? 0 : 16) + half * 8 + (v & 3) * 2 + (e & 1);
                af[mi][e] = As[am][kk];
            }
        }
        #pragma unroll
        for (int ni = 0; ni < 2; ++ni) {
            const int bn = nw * 32 + ni * 16 + l15;
            #pragma unroll
            for (int e = 0; e < 16; ++e) {
                bfm[ni][e] = Bs[((e < 8) ? 0 : 8) + half * 8 + e][bn];
            }
        }

        // ---- 4 WMMAs: 32x32 per wave per K-step ----
        #pragma unroll
        for (int mi = 0; mi < 2; ++mi)
            #pragma unroll
            for (int ni = 0; ni < 2; ++ni)
                acc[mi][ni] = __builtin_amdgcn_wmma_f32_16x16x32_bf16(
                    false, af[mi], false, bfm[ni], (short)0, acc[mi][ni],
                    false, false);
        __syncthreads();
    }

    // ---- epilogue: bias + ReLU, store bf16 and/or f32 ----
    #pragma unroll
    for (int mi = 0; mi < 2; ++mi) {
        #pragma unroll
        for (int ni = 0; ni < 2; ++ni) {
            const int ng = n0 + nw * 32 + ni * 16 + l15;
            #pragma unroll
            for (int v = 0; v < 8; ++v) {
                int mg = m0 + mw * 32 + mi * 16 + half * 8 + v;
                if (mg < Cout && ng < N) {
                    float val = acc[mi][ni][v] + (bias ? bias[mg] : 0.f);
                    if (relu && val < 0.f) val = 0.f;
                    size_t idx = (size_t)mg * N + ng;
                    if (outF) outF[idx] = val;
                    if (outB) outB[idx] = (bf16)val;
                }
            }
        }
    }
}

// ---------------------------------------------------------------------------
// MaxPool2d(2,2, ceil_mode=True)  (out-of-range taps are skipped == -inf pad)
// ---------------------------------------------------------------------------
__global__ void k_maxpool(const bf16* __restrict__ in, int C, int IH, int IW,
                          bf16* __restrict__ out, int OH, int OW)
{
    int i = blockIdx.x * blockDim.x + threadIdx.x;
    int total = C * OH * OW;
    if (i >= total) return;
    int ow = i % OW, oh = (i / OW) % OH, c = i / (OW * OH);
    float m = -INFINITY;
    #pragma unroll
    for (int dy = 0; dy < 2; ++dy)
        #pragma unroll
        for (int dx = 0; dx < 2; ++dx) {
            int iy = oh * 2 + dy, ix = ow * 2 + dx;
            if (iy < IH && ix < IW) {
                float v = (float)in[(size_t)c * IH * IW + iy * IW + ix];
                if (v > m) m = v;
            }
        }
    out[i] = (bf16)m;
}

// ---------------------------------------------------------------------------
// Plastic head: activ = activin @ (w + alpha*hebb) + 1000*onehot  (raw-reshape
// semantics of torch .view(-1,200)/.view(-1,6) reproduced exactly)
// ---------------------------------------------------------------------------
__global__ void k_plastic(const float* __restrict__ conv6,   // [200*81] NCHW flat
                          const float* __restrict__ w,       // [200,6]
                          const float* __restrict__ hebb,    // [200,6]
                          const float* __restrict__ alpha,   // [1]
                          const int*   __restrict__ y,       // [256*256]
                          float* __restrict__ activ)         // [81*6] (== [1,6,9,9] raw)
{
    int i = blockIdx.x * blockDim.x + threadIdx.x;
    if (i >= 81 * 6) return;
    int t = i / 6, k = i % 6;
    float a = alpha[0];
    float sum = 0.f;
    for (int j = 0; j < 200; ++j)
        sum += conv6[t * 200 + j] * (w[j * 6 + k] + a * hebb[j * 6 + k]);
    // onehot.reshape(-1,6)[t][k] comes from NCHW flat index t*6+k
    int f = t * 6 + k;
    int c = f / 81, p = f % 81;
    int py = p / 9, px = p % 9;
    int ih = (int)floorf((py + 0.5f) * 256.f / 9.f);
    int iw = (int)floorf((px + 0.5f) * 256.f / 9.f);
    if (y[ih * 256 + iw] == c) sum += 1000.f;
    activ[f] = sum;
}

// softmax over channel axis of the [1,6,9,9] raw view of activ
__global__ void k_softmax81(const float* __restrict__ activ, float* __restrict__ sm)
{
    int p = blockIdx.x * blockDim.x + threadIdx.x;
    if (p >= 81) return;
    float mx = -INFINITY;
    for (int c = 0; c < 6; ++c) { float v = activ[c * 81 + p]; if (v > mx) mx = v; }
    float e[6], s = 0.f;
    for (int c = 0; c < 6; ++c) { e[c] = expf(activ[c * 81 + p] - mx); s += e[c]; }
    for (int c = 0; c < 6; ++c) sm[c * 81 + p] = e[c] / s;
}

// hebb_new = (1-eta)^81 * hebb + sum_t eta*(1-eta)^(80-t) * activin[t]^T soft[t]
__global__ void k_hebb(const float* __restrict__ conv6,  // activin raw flat
                       const float* __restrict__ sm,     // [6,81] flat (soft raw view)
                       const float* __restrict__ hebb,
                       const float* __restrict__ eta_p,
                       float* __restrict__ out)          // [200*6]
{
    int i = blockIdx.x * blockDim.x + threadIdx.x;
    if (i >= 200 * 6) return;
    int j = i / 6, c = i % 6;
    float eta = eta_p[0], d = 1.f - eta;
    float accv = 0.f;
    for (int t = 0; t < 81; ++t) {
        float wt = eta * powf(d, (float)(80 - t));
        // soft.reshape(-1,6)[t][c] = sm flat index t*6+c
        accv += wt * conv6[t * 200 + j] * sm[t * 6 + c];
    }
    out[i] = powf(d, 81.f) * hebb[i] + accv;
}

// out[c,y,x] = a[c,y,x] + b[c, y+cy, x+cx]   (b is [C,BH,BW])
__global__ void k_addcrop(const float* __restrict__ a, const float* __restrict__ b,
                          int C, int H, int W, int BH, int BW, int cy, int cx,
                          float* __restrict__ out)
{
    int i = blockIdx.x * blockDim.x + threadIdx.x;
    int total = C * H * W;
    if (i >= total) return;
    int x = i % W, yv = (i / W) % H, c = i / (W * H);
    out[i] = a[i] + b[((size_t)c * BH + (yv + cy)) * BW + (x + cx)];
}

// ConvTranspose2d (weights [Cin,Cout,K,K], padding 0) with output crop fused:
// out[o,y,x] = full_deconv[o, y+cy, x+cx]
__global__ void k_deconv(const float* __restrict__ in, int Cin, int IH, int IW,
                         const float* __restrict__ wt, int Cout, int Kk, int S,
                         float* __restrict__ out, int OH, int OW, int cy, int cx)
{
    int i = blockIdx.x * blockDim.x + threadIdx.x;
    int total = Cout * OH * OW;
    if (i >= total) return;
    int x = i % OW, yv = (i / OW) % OH, o = i / (OW * OH);
    int yy = yv + cy, xx = x + cx;
    float accv = 0.f;
    for (int r = 0; r < Kk; ++r) {
        int ty = yy - r;
        if (ty < 0 || (ty % S)) continue;
        int iy = ty / S;
        if (iy >= IH) continue;
        for (int s = 0; s < Kk; ++s) {
            int tx = xx - s;
            if (tx < 0 || (tx % S)) continue;
            int ix = tx / S;
            if (ix >= IW) continue;
            for (int ci = 0; ci < Cin; ++ci)
                accv += in[((size_t)ci * IH + iy) * IW + ix] *
                        wt[(((size_t)ci * Cout + o) * Kk + r) * Kk + s];
        }
    }
    out[i] = accv;
}

// ---------------------------------------------------------------------------
// Host orchestration
// ---------------------------------------------------------------------------
struct CwDesc { int wi, bi, co, ci, k; };
static const CwDesc CW[18] = {
    { 6, 7,  64,   3, 3}, { 8, 9,  64,  64, 3},                 // b1
    {10,11, 128,  64, 3}, {12,13, 128, 128, 3},                 // b2
    {14,15, 256, 128, 3}, {16,17, 256, 256, 3}, {18,19, 256, 256, 3}, // b3
    {20,21, 512, 256, 3}, {22,23, 512, 512, 3}, {24,25, 512, 512, 3}, // b4
    {26,27, 512, 512, 3}, {28,29, 512, 512, 3}, {30,31, 512, 512, 3}, // b5
    {32,33,4096, 512, 7}, {34,35,4096,4096, 1}, {36,37, 200,4096, 1}, // b6
    {38,39,   6, 512, 1},                                       // sp4
    {40,41,   6, 256, 1},                                       // sp3
};

extern "C" void kernel_launch(void* const* d_in, const int* in_sizes, int n_in,
                              void* d_out, int out_size, void* d_ws, size_t ws_size,
                              hipStream_t stream)
{
    (void)in_sizes; (void)n_in; (void)out_size; (void)ws_size;
    char* wsb = (char*)d_ws;
    size_t off = 0;
    auto alloc = [&](size_t bytes) -> void* {
        void* p = wsb + off;
        off = (off + bytes + 255) & ~(size_t)255;
        return p;
    };

    // ---- bf16 weights ----
    bf16* wbf[18];
    for (int i = 0; i < 18; ++i) {
        size_t n = (size_t)CW[i].co * CW[i].ci * CW[i].k * CW[i].k;
        wbf[i] = (bf16*)alloc(n * sizeof(bf16));
    }
    // ---- feature buffers ----
    bf16* xb = (bf16*)alloc((size_t)3 * 256 * 256 * sizeof(bf16));
    const size_t TBIG = (size_t)64 * 454 * 454;         // largest conv output
    bf16* tA = (bf16*)alloc(TBIG * sizeof(bf16));
    bf16* tB = (bf16*)alloc(TBIG * sizeof(bf16));
    bf16* p1 = (bf16*)alloc((size_t)64  * 227 * 227 * sizeof(bf16));
    bf16* p2 = (bf16*)alloc((size_t)128 * 114 * 114 * sizeof(bf16));
    bf16* p3 = (bf16*)alloc((size_t)256 *  57 *  57 * sizeof(bf16));
    bf16* p4 = (bf16*)alloc((size_t)512 *  29 *  29 * sizeof(bf16));
    bf16* p5 = (bf16*)alloc((size_t)512 *  15 *  15 * sizeof(bf16));
    bf16* f6 = (bf16*)alloc((size_t)4096 * 81 * sizeof(bf16));
    bf16* f7 = (bf16*)alloc((size_t)4096 * 81 * sizeof(bf16));
    float* conv6f = (float*)alloc((size_t)200 * 81 * sizeof(float));
    float* activ  = (float*)alloc(486 * sizeof(float));   // == activout [1,6,9,9] raw
    float* smbuf  = (float*)alloc(486 * sizeof(float));
    float* sp4f   = (float*)alloc((size_t)6 * 29 * 29 * sizeof(float));
    float* sp3f   = (float*)alloc((size_t)6 * 57 * 57 * sizeof(float));
    float* u2     = (float*)alloc((size_t)6 * 20 * 20 * sizeof(float));
    float* fi4    = (float*)alloc((size_t)6 * 20 * 20 * sizeof(float));
    float* u4     = (float*)alloc((size_t)6 * 42 * 42 * sizeof(float));
    float* fi8    = (float*)alloc((size_t)6 * 42 * 42 * sizeof(float));

    auto g1 = [](size_t n) { return dim3((unsigned)((n + 255) / 256)); };

    // ---- convert weights + input to bf16 ----
    for (int i = 0; i < 18; ++i) {
        size_t n = (size_t)CW[i].co * CW[i].ci * CW[i].k * CW[i].k;
        k_cvt_bf16<<<g1(n), 256, 0, stream>>>((const float*)d_in[CW[i].wi], wbf[i], (int)n);
    }
    k_cvt_bf16<<<g1(3 * 256 * 256), 256, 0, stream>>>((const float*)d_in[0], xb, 3 * 256 * 256);

    auto conv = [&](const bf16* in, int Cin, int IH, int IW, int li, int pad,
                    bf16* oB, float* oF, int relu, int OH, int OW) {
        dim3 grid((OH * OW + BN - 1) / BN, (CW[li].co + BM - 1) / BM);
        k_conv_wmma<<<grid, 256, 0, stream>>>(in, Cin, IH, IW, wbf[li],
                                              (const float*)d_in[CW[li].bi],
                                              CW[li].co, CW[li].k, CW[li].k, pad,
                                              oB, oF, OH, OW, relu);
    };
    auto pool = [&](const bf16* in, int C, int IH, int IW, bf16* o, int OH, int OW) {
        k_maxpool<<<g1((size_t)C * OH * OW), 256, 0, stream>>>(in, C, IH, IW, o, OH, OW);
    };

    // ---- VGG trunk ----
    conv(xb,   3, 256, 256,  0, 100, tA, nullptr, 1, 454, 454);   // conv1_1 (pad 100)
    conv(tA,  64, 454, 454,  1,   1, tB, nullptr, 1, 454, 454);
    pool(tB,  64, 454, 454, p1, 227, 227);
    conv(p1,  64, 227, 227,  2,   1, tA, nullptr, 1, 227, 227);
    conv(tA, 128, 227, 227,  3,   1, tB, nullptr, 1, 227, 227);
    pool(tB, 128, 227, 227, p2, 114, 114);
    conv(p2, 128, 114, 114,  4,   1, tA, nullptr, 1, 114, 114);
    conv(tA, 256, 114, 114,  5,   1, tB, nullptr, 1, 114, 114);
    conv(tB, 256, 114, 114,  6,   1, tA, nullptr, 1, 114, 114);
    pool(tA, 256, 114, 114, p3,  57,  57);                         // c3
    conv(p3, 256,  57,  57,  7,   1, tA, nullptr, 1,  57,  57);
    conv(tA, 512,  57,  57,  8,   1, tB, nullptr, 1,  57,  57);
    conv(tB, 512,  57,  57,  9,   1, tA, nullptr, 1,  57,  57);
    pool(tA, 512,  57,  57, p4,  29,  29);                         // c4
    conv(p4, 512,  29,  29, 10,   1, tA, nullptr, 1,  29,  29);
    conv(tA, 512,  29,  29, 11,   1, tB, nullptr, 1,  29,  29);
    conv(tB, 512,  29,  29, 12,   1, tA, nullptr, 1,  29,  29);
    pool(tA, 512,  29,  29, p5,  15,  15);                         // c5

    // ---- fc6 / fc7 / score200 ----
    conv(p5,  512, 15, 15, 13, 0, f6, nullptr, 1, 9, 9);           // 7x7 conv, ReLU
    conv(f6, 4096,  9,  9, 14, 0, f7, nullptr, 1, 9, 9);           // 1x1 conv, ReLU
    conv(f7, 4096,  9,  9, 15, 0, nullptr, conv6f, 0, 9, 9);       // score -> f32

    // ---- plastic head ----
    k_plastic<<<g1(486), 256, 0, stream>>>(conv6f, (const float*)d_in[3],
                                           (const float*)d_in[2], (const float*)d_in[4],
                                           (const int*)d_in[1], activ);
    k_softmax81<<<1, 128, 0, stream>>>(activ, smbuf);
    k_hebb<<<g1(1200), 256, 0, stream>>>(conv6f, smbuf, (const float*)d_in[2],
                                         (const float*)d_in[5],
                                         (float*)d_out + 6 * 256 * 256);

    // ---- score pools (1x1 convs via WMMA path) ----
    conv(p4, 512, 29, 29, 16, 0, nullptr, sp4f, 0, 29, 29);        // sp4
    conv(p3, 256, 57, 57, 17, 0, nullptr, sp3f, 0, 57, 57);        // sp3

    // ---- FCN8s decoder ----
    k_deconv<<<g1((size_t)6 * 20 * 20), 256, 0, stream>>>(
        activ, 6, 9, 9, (const float*)d_in[42], 6, 4, 2, u2, 20, 20, 0, 0);
    k_addcrop<<<g1((size_t)6 * 20 * 20), 256, 0, stream>>>(
        u2, sp4f, 6, 20, 20, 29, 29, 5, 5, fi4);
    k_deconv<<<g1((size_t)6 * 42 * 42), 256, 0, stream>>>(
        fi4, 6, 20, 20, (const float*)d_in[43], 6, 4, 2, u4, 42, 42, 0, 0);
    k_addcrop<<<g1((size_t)6 * 42 * 42), 256, 0, stream>>>(
        u4, sp3f, 6, 42, 42, 57, 57, 9, 9, fi8);
    k_deconv<<<g1((size_t)6 * 256 * 256), 256, 0, stream>>>(
        fi8, 6, 42, 42, (const float*)d_in[44], 6, 16, 8,
        (float*)d_out, 256, 256, 31, 31);
}
